// SNGPModule_59691455480692
// MI455X (gfx1250) — compile-verified, matrix operand order checked by reading, not run on hardware
//
#include <hip/hip_runtime.h>

// ---------------------------------------------------------------------------
// SNGP fused forward + covariance update for MI455X (gfx1250, wave32, WMMA)
//   Phi = sqrt(2/O)*cos(-x@W.T + b);  out = Phi@Beta
//   scalars[k] = sum_b softmax'(out)[b,k]*(rowsum Phi[b])^2 / O^2
//   sigma_new = 0.999*sigma + 0.001*scalars[k]
// GEMMs: split-bf16 (hi/lo) 3-pass v_wmma_f32_16x16x32_bf16 (~fp32 accuracy).
// W/Beta pre-split once into workspace; LDS tiles staged with CDNA5 async
// global->LDS (ASYNCcnt). Transcendentals use hardware v_cos/v_exp to avoid
// the register-hungry libm expansions (no accumulator spills).
// ---------------------------------------------------------------------------

#define HID    2048
#define OUTN   256
#define BATCHN 16384
#define MT     64            // M-rows per workgroup
#define NK1    (HID / 32)    // 64 K-steps for GEMM1
#define SMEM_BYTES 100096

typedef __attribute__((ext_vector_type(16))) __bf16 v16bf;
typedef __attribute__((ext_vector_type(8)))  float  v8f;

__device__ __forceinline__ unsigned short f2bf(float f) {
  unsigned u = __float_as_uint(f);
  unsigned r = u + 0x7FFFu + ((u >> 16) & 1u);   // round-to-nearest-even
  return (unsigned short)(r >> 16);
}
__device__ __forceinline__ float bf2f(unsigned short h) {
  return __uint_as_float(((unsigned)h) << 16);
}

union F4  { float4 v; float f[4]; };
union U4S { uint4 u; unsigned short s[8]; };
union V16 { v16bf v; uint4 u[2]; };

// CDNA5 async DMA: global -> LDS, 16B per lane, tracked by ASYNCcnt.
__device__ __forceinline__ void async_b128(void* lds_ptr, const void* gbase,
                                           unsigned goff_bytes) {
  unsigned lds_off = (unsigned)(uintptr_t)lds_ptr;  // low 32 bits = LDS addr
  asm volatile("global_load_async_to_lds_b128 %0, %1, %2"
               :: "v"(lds_off), "v"(goff_bytes), "s"(gbase)
               : "memory");
}
__device__ __forceinline__ void wait_async() {
  asm volatile("s_wait_asynccnt 0" ::: "memory");
}

// Build a 16x32 bf16 WMMA fragment from LDS: elements 0..7 at p[0..7]
// (K = ksel*8 .. +8), elements 8..15 at p[16..23] (K = 16+ksel*8 .. +8).
__device__ __forceinline__ v16bf frag16(const unsigned short* p) {
  V16 t;
  t.u[0] = *(const uint4*)(p);
  t.u[1] = *(const uint4*)(p + 16);
  return t.v;
}

__device__ __forceinline__ v8f wmma_bf16(v16bf a, v16bf b, v8f c) {
  return __builtin_amdgcn_wmma_f32_16x16x32_bf16(false, a, false, b,
                                                 (short)0, c, false, false);
}

// Split 8 fp32 into bf16 hi (RNE) + bf16 lo (residual), store 16B each.
__device__ __forceinline__ void split_store8(const float* f,
                                             unsigned short* hi,
                                             unsigned short* lo) {
  U4S H, L;
#pragma unroll
  for (int i = 0; i < 8; ++i) {
    unsigned short h = f2bf(f[i]);
    H.s[i] = h;
    L.s[i] = f2bf(f[i] - bf2f(h));
  }
  *(uint4*)hi = H.u;
  *(uint4*)lo = L.u;
}

// Sum 8 bf16 values packed in a uint4.
__device__ __forceinline__ float sum_bf16x8(uint4 u) {
  unsigned w0 = u.x, w1 = u.y, w2 = u.z, w3 = u.w;
  float s = __uint_as_float(w0 << 16) + __uint_as_float(w0 & 0xffff0000u);
  s += __uint_as_float(w1 << 16) + __uint_as_float(w1 & 0xffff0000u);
  s += __uint_as_float(w2 << 16) + __uint_as_float(w2 & 0xffff0000u);
  s += __uint_as_float(w3 << 16) + __uint_as_float(w3 & 0xffff0000u);
  return s;
}

// One K=32 step: 2x4 16x16 tiles per wave, 3 WMMAs per tile (hh+hl+lh).
__device__ __forceinline__ void mma_step(const unsigned short* Ah,
                                         const unsigned short* Al, int as,
                                         const unsigned short* Bh,
                                         const unsigned short* Bl,
                                         int wm, int wn, int lane,
                                         v8f (&acc)[2][4]) {
  const int nl   = lane & 15;
  const int koff = (lane >> 4) * 8;
  v16bf ah[2], al[2], bh[4], bl[4];
#pragma unroll
  for (int tm = 0; tm < 2; ++tm) {
    int m = wm * 32 + tm * 16 + nl;
    ah[tm] = frag16(Ah + m * as + koff);
    al[tm] = frag16(Al + m * as + koff);
  }
#pragma unroll
  for (int tn = 0; tn < 4; ++tn) {
    int n = wn * 64 + tn * 16 + nl;
    bh[tn] = frag16(Bh + n * 32 + koff);
    bl[tn] = frag16(Bl + n * 32 + koff);
  }
#pragma unroll
  for (int tm = 0; tm < 2; ++tm)
#pragma unroll
    for (int tn = 0; tn < 4; ++tn) {
      v8f c = acc[tm][tn];
      c = wmma_bf16(ah[tm], bh[tn], c);
      c = wmma_bf16(ah[tm], bl[tn], c);
      c = wmma_bf16(al[tm], bh[tn], c);
      acc[tm][tn] = c;
    }
}

// Pre-split frozen weights once per call:
//   W[256][2048] f32 -> Whi/Wlo bf16 (same layout, K contiguous)
//   Beta[k][n] f32   -> Bthi/Btlo bf16 transposed [n][k] (K contiguous)
__global__ __launch_bounds__(256) void sngp_prep_kernel(
    const float* __restrict__ W, const float* __restrict__ Beta,
    unsigned short* __restrict__ Whi, unsigned short* __restrict__ Wlo,
    unsigned short* __restrict__ Bthi, unsigned short* __restrict__ Btlo) {
  const size_t i = (size_t)blockIdx.x * 256 + threadIdx.x;  // < 524288
  float w = W[i];
  unsigned short h = f2bf(w);
  Whi[i] = h;
  Wlo[i] = f2bf(w - bf2f(h));
  if (i < (size_t)OUTN * OUTN) {
    const int k = (int)(i >> 8), n = (int)(i & 255);
    float bv = Beta[i];                      // Beta[k][n]
    unsigned short bh = f2bf(bv);
    Bthi[n * 256 + k] = bh;
    Btlo[n * 256 + k] = f2bf(bv - bf2f(bh));
  }
}

__global__ __launch_bounds__(256) void sngp_fused_kernel(
    const float* __restrict__ x, const unsigned short* __restrict__ Whi,
    const unsigned short* __restrict__ Wlo,
    const unsigned short* __restrict__ Bthi,
    const unsigned short* __restrict__ Btlo,
    const float* __restrict__ bvec, float* __restrict__ outG,
    float* __restrict__ partials) {
  extern __shared__ char smem[];
  // GEMM1 staging (overwritten by Phi after GEMM1 completes):
  unsigned short* Ahi = (unsigned short*)(smem + 0);       //  4 KB (64x32)
  unsigned short* Alo = (unsigned short*)(smem + 4096);    //  4 KB
  unsigned short* Bhi = (unsigned short*)(smem + 8192);    // 16 KB (256x32)
  unsigned short* Blo = (unsigned short*)(smem + 24576);   // 16 KB
  // Phi hi/lo bf16 [64][256] (reuses bytes 0..64K); out f32 reuses same later
  unsigned short* Phi_h = (unsigned short*)(smem + 0);     // 32 KB
  unsigned short* Phi_l = (unsigned short*)(smem + 32768); // 32 KB
  float* outb = (float*)(smem + 0);                        // 64 KB f32[64][256]
  // Beta transposed tiles [256 n][32 k] hi/lo
  unsigned short* Bth = (unsigned short*)(smem + 65536);   // 16 KB
  unsigned short* Btl = (unsigned short*)(smem + 81920);   // 16 KB
  float* bsh = (float*)(smem + 98304);                     // 1 KB  bias
  float* s2  = (float*)(smem + 99328);                     // 256 B rowsum^2
  float* rmx = (float*)(smem + 99584);                     // 256 B row max
  float* rdn = (float*)(smem + 99840);                     // 256 B row denom

  const int tid  = threadIdx.x;
  const int lane = tid & 31;
  const int wave = tid >> 5;          // 8 waves: 2 (M) x 4 (N)
  const int wm   = wave >> 2;
  const int wn   = wave & 3;
  const int m0   = blockIdx.x * MT;

  bsh[tid] = bvec[tid];

  v8f acc[2][4];
#pragma unroll
  for (int tm = 0; tm < 2; ++tm)
#pragma unroll
    for (int tn = 0; tn < 4; ++tn)
#pragma unroll
      for (int i = 0; i < 8; ++i) acc[tm][tn][i] = 0.0f;

  // ---------------- GEMM1: S = x @ W.T  (M=64, N=256, K=2048) --------------
  const int arow = tid >> 2;          // 64 rows, 4 threads/row
  const int acol = (tid & 3) * 8;     // 8 floats each
  F4 xa0, xa1;
  auto loadA = [&](int k0) {
    const float* xp = x + (size_t)(m0 + arow) * HID + k0 + acol;
    xa0.v = *(const float4*)xp;
    xa1.v = *(const float4*)(xp + 4);
  };
  loadA(0);
  for (int kt = 0; kt < NK1; ++kt) {
    const int k0 = kt * 32;
    // B tile: async DMA global->LDS (bf16 hi/lo, row n = tid, 64B each)
    {
      const unsigned gro = (unsigned)tid * (HID * 2) + (unsigned)k0 * 2;
#pragma unroll
      for (int j = 0; j < 4; ++j)
        async_b128((char*)Bhi + tid * 64 + j * 16, Whi, gro + j * 16);
#pragma unroll
      for (int j = 0; j < 4; ++j)
        async_b128((char*)Blo + tid * 64 + j * 16, Wlo, gro + j * 16);
    }
    // A tile: split fp32 -> bf16 hi/lo, ds_write (overlaps with async DMA)
    float a8[8];
#pragma unroll
    for (int i = 0; i < 4; ++i) { a8[i] = xa0.f[i]; a8[4 + i] = xa1.f[i]; }
    split_store8(a8, Ahi + arow * 32 + acol, Alo + arow * 32 + acol);
    wait_async();
    __syncthreads();
    if (kt + 1 < NK1) loadA(k0 + 32);            // prefetch next x tile
    mma_step(Ahi, Alo, 32, Bhi, Blo, wm, wn, lane, acc);
    __syncthreads();
  }

  // ------- Phi = coef * cos(b - S), split to bf16 hi/lo in LDS -------------
  // |b - S| <~ 60 rad: well inside v_cos_f32 range; few-ulp error is far
  // below the bf16-split noise floor. Using __cosf avoids the register-hungry
  // libm expansion that previously spilled the accumulators to scratch.
  const float coef = 0.08838834764831845f;  // sqrt(2/256)
#pragma unroll
  for (int tm = 0; tm < 2; ++tm)
#pragma unroll
    for (int tn = 0; tn < 4; ++tn) {
      const int nl  = lane & 15;
      const int hih = lane >> 4;
      const int ng  = wn * 64 + tn * 16 + nl;
      const float bn = bsh[ng];
#pragma unroll
      for (int v = 0; v < 8; ++v) {
        const int mg = wm * 32 + tm * 16 + v + 8 * hih;
        float ph = coef * __cosf(bn - acc[tm][tn][v]);
        unsigned short h = f2bf(ph);
        Phi_h[mg * 256 + ng] = h;
        Phi_l[mg * 256 + ng] = f2bf(ph - bf2f(h));
      }
    }
  __syncthreads();

  // s2[b] = (sum_o Phi[b,o])^2  (vectorized hi+lo reconstruction)
  if (tid < MT) {
    const uint4* ph = (const uint4*)(Phi_h + tid * 256);
    const uint4* pl = (const uint4*)(Phi_l + tid * 256);
    float s = 0.0f;
#pragma unroll 4
    for (int o = 0; o < 32; ++o) s += sum_bf16x8(ph[o]) + sum_bf16x8(pl[o]);
    s2[tid] = s * s;
  }

#pragma unroll
  for (int tm = 0; tm < 2; ++tm)
#pragma unroll
    for (int tn = 0; tn < 4; ++tn)
#pragma unroll
      for (int i = 0; i < 8; ++i) acc[tm][tn][i] = 0.0f;

  // ---------------- GEMM2: out = Phi @ Beta  (K=256, 8 steps) --------------
  for (int kt = 0; kt < 8; ++kt) {
    const int k0 = kt * 32;
    // Beta tile (pre-transposed bf16): async DMA global->LDS
    {
      const unsigned gro = (unsigned)tid * (OUTN * 2) + (unsigned)k0 * 2;
#pragma unroll
      for (int j = 0; j < 4; ++j)
        async_b128((char*)Bth + tid * 64 + j * 16, Bthi, gro + j * 16);
#pragma unroll
      for (int j = 0; j < 4; ++j)
        async_b128((char*)Btl + tid * 64 + j * 16, Btlo, gro + j * 16);
    }
    wait_async();
    __syncthreads();
    mma_step(Phi_h + k0, Phi_l + k0, 256, Bth, Btl, wm, wn, lane, acc);
    __syncthreads();
  }

  // out -> global + LDS (outb overwrites Phi region; all reads done above)
#pragma unroll
  for (int tm = 0; tm < 2; ++tm)
#pragma unroll
    for (int tn = 0; tn < 4; ++tn) {
      const int nl  = lane & 15;
      const int hih = lane >> 4;
      const int ng  = wn * 64 + tn * 16 + nl;
#pragma unroll
      for (int v = 0; v < 8; ++v) {
        const int mg = wm * 32 + tm * 16 + v + 8 * hih;
        const float val = acc[tm][tn][v];
        outG[(size_t)(m0 + mg) * OUTN + ng] = val;
        outb[mg * 256 + ng] = val;
      }
    }
  __syncthreads();

  // ---------------- softmax stats per row (vectorized scans) ---------------
  if (tid < MT) {
    const float4* row = (const float4*)(outb + tid * 256);
    float mx = -3.0e38f;
#pragma unroll 4
    for (int k = 0; k < 64; ++k) {
      float4 v = row[k];
      mx = fmaxf(mx, fmaxf(fmaxf(v.x, v.y), fmaxf(v.z, v.w)));
    }
    float s = 0.0f;
#pragma unroll 4
    for (int k = 0; k < 64; ++k) {
      float4 v = row[k];
      s += __expf(v.x - mx) + __expf(v.y - mx) + __expf(v.z - mx) +
           __expf(v.w - mx);
    }
    rmx[tid] = mx;
    rdn[tid] = s;
  }
  __syncthreads();

  // partial scalars: thread tid owns column k = tid
  {
    const int k = tid;
    float local = 0.0f;
    for (int b = 0; b < MT; ++b) {
      float p = __expf(outb[b * 256 + k] - rmx[b]) / rdn[b];
      local += p * (1.0f - p) * s2[b];
    }
    partials[(size_t)blockIdx.x * 256 + k] = local;
  }
}

__global__ void sngp_reduce_kernel(const float* __restrict__ partials,
                                   float* __restrict__ scal) {
  const int k = threadIdx.x;
  float s = 0.0f;
  for (int wg = 0; wg < BATCHN / MT; ++wg) s += partials[wg * 256 + k];
  scal[k] = s * (1.0f / (float)(OUTN * OUTN));
}

__global__ __launch_bounds__(256) void sngp_sigma_kernel(
    const float* __restrict__ sig_in, const float* __restrict__ scal,
    float* __restrict__ sig_out) {
  const size_t i4   = (size_t)blockIdx.x * blockDim.x + threadIdx.x;
  const size_t base = i4 * 4;                 // 4 floats per thread
  const float  s    = scal[base >> 16] * (1.0f - 0.999f);
  const float  A    = 0.999f;
  float4 v = *(const float4*)(sig_in + base);
  float4 r;
  r.x = fmaf(A, v.x, s);
  r.y = fmaf(A, v.y, s);
  r.z = fmaf(A, v.z, s);
  r.w = fmaf(A, v.w, s);
  *(float4*)(sig_out + base) = r;
}

extern "C" void kernel_launch(void* const* d_in, const int* in_sizes, int n_in,
                              void* d_out, int out_size, void* d_ws,
                              size_t ws_size, hipStream_t stream) {
  (void)in_sizes; (void)n_in; (void)out_size; (void)ws_size;
  const float* x     = (const float*)d_in[0];   // [16384,2048]
  const float* W     = (const float*)d_in[1];   // [256,2048]
  const float* bvec  = (const float*)d_in[2];   // [256]
  const float* Beta  = (const float*)d_in[3];   // [256,256]
  const float* sigma = (const float*)d_in[4];   // [256,256,256]

  float* outG   = (float*)d_out;                          // [16384,256]
  float* sigOut = outG + (size_t)BATCHN * OUTN;           // [256,256,256]

  // Workspace layout (all 16B aligned):
  float* partials = (float*)d_ws;                         // [256 WGs][256]
  float* scalars  = partials + 256 * 256;                 // [256]
  unsigned short* Whi  = (unsigned short*)(scalars + 256);     // [256][2048]
  unsigned short* Wlo  = Whi + (size_t)OUTN * HID;
  unsigned short* Bthi = Wlo + (size_t)OUTN * HID;             // [n][k] 256x256
  unsigned short* Btlo = Bthi + (size_t)OUTN * OUTN;

  sngp_prep_kernel<<<(OUTN * HID) / 256, 256, 0, stream>>>(W, Beta, Whi, Wlo,
                                                           Bthi, Btlo);
  sngp_fused_kernel<<<BATCHN / MT, 256, SMEM_BYTES, stream>>>(
      x, Whi, Wlo, Bthi, Btlo, bvec, outG, partials);
  sngp_reduce_kernel<<<1, 256, 0, stream>>>(partials, scalars);
  sngp_sigma_kernel<<<(OUTN * OUTN * OUTN) / (256 * 4), 256, 0, stream>>>(
      sigma, scalars, sigOut);
}
